// RegionGridAggregator_79482664780153
// MI455X (gfx1250) — matrix operand order repaired and checked by reading, not compiled, not run on hardware
//
#include <hip/hip_runtime.h>

// ---------------- problem constants ----------------
#define BB    256
#define MM    576
#define PP    64
#define KC    10
#define NN    650
#define NP    672          // padded rows (multiple of 32 for 32-row GEMM blocking)
#define DD    256
#define HH    8
#define DHH   32
#define GSZ   24
#define EMAXE 6560         // 2208 grid + 640 sim + 3456 corr + 256 att

typedef __attribute__((ext_vector_type(16))) __bf16 v16bf;
typedef __attribute__((ext_vector_type(8)))  __bf16 v8bf;
typedef __attribute__((ext_vector_type(8)))  float  v8f;

// ---------------- helpers ----------------
__device__ inline unsigned short f2bf(float f) {
  unsigned u = __float_as_uint(f);
  u += 0x7FFFu + ((u >> 16) & 1u);           // round-to-nearest-even
  return (unsigned short)(u >> 16);
}
__device__ inline float bf2f(unsigned short h) { return __uint_as_float(((unsigned)h) << 16); }
// order-preserving float<->uint for atomic max
__device__ inline unsigned fenc(float f) {
  unsigned u = __float_as_uint(f);
  return (u & 0x80000000u) ? ~u : (u | 0x80000000u);
}
__device__ inline float fdec(unsigned u) {
  return __uint_as_float((u & 0x80000000u) ? (u & 0x7FFFFFFFu) : ~u);
}
__device__ inline v16bf load_afrag(const unsigned short* p, int ka) {
  v8bf x0 = *reinterpret_cast<const v8bf*>(p + ka);
  v8bf x1 = *reinterpret_cast<const v8bf*>(p + ka + 16);
  v16bf r;
#pragma unroll
  for (int i = 0; i < 8; ++i) { r[i] = x0[i]; r[i + 8] = x1[i]; }
  return r;
}

// ---------------- register-blocked bf16 WMMA GEMM ----------------
// C[b][m][n] = sum_k A[b][m][k] * Bt[n][k] + bias[n]
// A: bf16 row-major [*, lda]; Bt: bf16 row-major [Ncols, ldb] (transposed weight)
// One wave computes a 32(M) x 64(N) block: 2 A-frags x 4 B-frags -> 8 WMMA / K-step.
__global__ __launch_bounds__(256)
void gemm_bf16_wmma(const unsigned short* __restrict__ A, long sA, int lda,
                    const unsigned short* __restrict__ Bt, int ldb,
                    const float* __restrict__ bias,
                    float* __restrict__ Cf, long sCf,
                    unsigned short* __restrict__ Cbf, long sCbf,
                    int ldc, int blocksM, int blocksN, int Kd)
{
  const int wave = threadIdx.x >> 5;
  const int lane = threadIdx.x & 31;
  const int blk  = blockIdx.x * (blockDim.x >> 5) + wave;
  if (blk >= blocksM * blocksN) return;
  const int bm = blk / blocksN, bn = blk % blocksN;
  const int b  = blockIdx.y;
  const int r16   = lane & 15;
  const int khalf = lane >> 4;

  const unsigned short* Ar0 = A + (size_t)b * sA + (size_t)(bm * 32 + r16) * lda;
  const unsigned short* Ar1 = Ar0 + (size_t)16 * lda;
  const unsigned short* Br0 = Bt + (size_t)(bn * 64 + r16) * ldb;
  const unsigned short* Br1 = Br0 + (size_t)16 * ldb;
  const unsigned short* Br2 = Br0 + (size_t)32 * ldb;
  const unsigned short* Br3 = Br0 + (size_t)48 * ldb;

  v8f acc[2][4] = {};
  for (int kt = 0; kt < Kd; kt += 32) {
    // A fragment: lane<16 -> K {kt..kt+7, kt+16..kt+23}; lane>=16 -> +8
    const int ka = kt + khalf * 8;
    const v16bf a0 = load_afrag(Ar0, ka);
    const v16bf a1 = load_afrag(Ar1, ka);
    // B fragment: same K order as A (lane<16 -> K {kt..kt+7, kt+16..kt+23}; lane>=16 -> +8)
    const int kb = kt + khalf * 8;
    v16bf bf0 = load_afrag(Br0, kb);
    v16bf bf1 = load_afrag(Br1, kb);
    v16bf bf2 = load_afrag(Br2, kb);
    v16bf bf3 = load_afrag(Br3, kb);
    if (kt + 32 < Kd) {                       // gfx1250 global_prefetch_b8
      __builtin_prefetch(Ar0 + kt + 32, 0, 0);
      __builtin_prefetch(Br0 + kt + 32, 0, 0);
    }
    acc[0][0] = __builtin_amdgcn_wmma_f32_16x16x32_bf16(false, a0, false, bf0, (short)0, acc[0][0], false, false);
    acc[0][1] = __builtin_amdgcn_wmma_f32_16x16x32_bf16(false, a0, false, bf1, (short)0, acc[0][1], false, false);
    acc[0][2] = __builtin_amdgcn_wmma_f32_16x16x32_bf16(false, a0, false, bf2, (short)0, acc[0][2], false, false);
    acc[0][3] = __builtin_amdgcn_wmma_f32_16x16x32_bf16(false, a0, false, bf3, (short)0, acc[0][3], false, false);
    acc[1][0] = __builtin_amdgcn_wmma_f32_16x16x32_bf16(false, a1, false, bf0, (short)0, acc[1][0], false, false);
    acc[1][1] = __builtin_amdgcn_wmma_f32_16x16x32_bf16(false, a1, false, bf1, (short)0, acc[1][1], false, false);
    acc[1][2] = __builtin_amdgcn_wmma_f32_16x16x32_bf16(false, a1, false, bf2, (short)0, acc[1][2], false, false);
    acc[1][3] = __builtin_amdgcn_wmma_f32_16x16x32_bf16(false, a1, false, bf3, (short)0, acc[1][3], false, false);
  }

  float* Cfb = Cf ? Cf + (size_t)b * sCf : nullptr;
  unsigned short* Cbb = Cbf ? Cbf + (size_t)b * sCbf : nullptr;
#pragma unroll
  for (int i2 = 0; i2 < 2; ++i2) {
#pragma unroll
    for (int j = 0; j < 4; ++j) {
      const int n = bn * 64 + j * 16 + r16;
      const float bv = bias ? bias[n] : 0.f;
#pragma unroll
      for (int i = 0; i < 8; ++i) {
        const int m = bm * 32 + i2 * 16 + khalf * 8 + i;  // D: lanes>=16 hold rows 8..15
        const float v = acc[i2][j][i] + bv;
        const size_t off = (size_t)m * ldc + n;
        if (Cfb) Cfb[off] = v;
        if (Cbb) Cbb[off] = f2bf(v);
      }
    }
  }
}

// ---------------- conversion / packing ----------------
__global__ void f2bf_kernel(const float* __restrict__ in, unsigned short* __restrict__ out, long n) {
  long i = (long)blockIdx.x * blockDim.x + threadIdx.x;
  long st = (long)gridDim.x * blockDim.x;
  for (; i < n; i += st) out[i] = f2bf(in[i]);
}
// Wt[n*K + k] = bf16(W[k*N + n])
__global__ void pack_wt_kernel(const float* __restrict__ W, unsigned short* __restrict__ Wt,
                               int K, int N) {
  int i = blockIdx.x * blockDim.x + threadIdx.x;
  if (i >= K * N) return;
  int k = i / N, n = i % N;
  Wt[(size_t)n * K + k] = f2bf(W[i]);
}
__global__ void copy_f32_kernel(const float* __restrict__ in, float* __restrict__ out, int n) {
  int i = blockIdx.x * blockDim.x + threadIdx.x;
  if (i < n) out[i] = in[i];
}
__global__ void zero_pad_kernel(float* __restrict__ x, unsigned short* __restrict__ xbf) {
  int i = blockIdx.x * blockDim.x + threadIdx.x;   // BB * (NP-NN) * 256
  if (i >= BB * (NP - NN) * 256) return;
  int c = i % 256, rr = (i / 256) % (NP - NN), b = i / (256 * (NP - NN));
  size_t off = ((size_t)b * NP + NN + rr) * DD + c;
  x[off] = 0.f; xbf[off] = 0;
}
__global__ void init_att_kernel(unsigned* __restrict__ smax, float* __restrict__ den,
                                float* __restrict__ hagg) {
  long i = (long)blockIdx.x * blockDim.x + threadIdx.x;
  long nh = (long)BB * NP * HH;
  long nc = (long)BB * NP * DD;
  long st = (long)gridDim.x * blockDim.x;
  for (long j = i; j < nc; j += st) {
    if (j < nh) { smax[j] = fenc(-1.0e30f); den[j] = 0.f; }
    hagg[j] = 0.f;
  }
}

// ---------------- cluster softmax aggregation ----------------
// corr0 = softmax(gh @ centers^T + bias); agg[k] = sum_m corr0[m,k]*(gh[m]-rbar); l2norm
__global__ __launch_bounds__(256)
void cluster_agg_kernel(float* __restrict__ x, unsigned short* __restrict__ xbf,
                        const float* __restrict__ centers, const float* __restrict__ rbar,
                        const float* __restrict__ gbias, const float* __restrict__ rbias)
{
  const int b = blockIdx.x;
  const float* g = x + (size_t)b * NP * DD;          // rows 0..575 = gh
  __shared__ float aggS[KC * DD];
  for (int i = threadIdx.x; i < KC * DD; i += blockDim.x) aggS[i] = 0.f;
  __syncthreads();
  const int w = threadIdx.x >> 5, lane = threadIdx.x & 31;
  const float bias = gbias[0] + rbias[0];
  for (int m = w; m < MM; m += 8) {
    float sc[KC];
#pragma unroll
    for (int k = 0; k < KC; ++k) {
      float s = 0.f;
#pragma unroll
      for (int cc = 0; cc < 8; ++cc) {
        int c = lane + 32 * cc;
        s += g[m * DD + c] * centers[k * DD + c];
      }
      for (int o = 16; o; o >>= 1) s += __shfl_xor(s, o);
      sc[k] = s + bias;
    }
    float mx = sc[0];
#pragma unroll
    for (int k = 1; k < KC; ++k) mx = fmaxf(mx, sc[k]);
    float den = 0.f;
#pragma unroll
    for (int k = 0; k < KC; ++k) { sc[k] = __expf(sc[k] - mx); den += sc[k]; }
#pragma unroll
    for (int k = 0; k < KC; ++k) {
      float wk = sc[k] / den;
#pragma unroll
      for (int cc = 0; cc < 8; ++cc) {
        int c = lane + 32 * cc;
        atomicAdd(&aggS[k * DD + c], wk * (g[m * DD + c] - rbar[c]));  // ds_add_f32
      }
    }
  }
  __syncthreads();
  for (int k = w; k < KC; k += 8) {
    float s = 0.f;
#pragma unroll
    for (int cc = 0; cc < 8; ++cc) { float v = aggS[k * DD + lane + 32 * cc]; s += v * v; }
    for (int o = 16; o; o >>= 1) s += __shfl_xor(s, o);
    float inv = 1.f / fmaxf(sqrtf(s), 1e-12f);
#pragma unroll
    for (int cc = 0; cc < 8; ++cc) {
      int c = lane + 32 * cc;
      float v = aggS[k * DD + c] * inv;
      size_t off = ((size_t)b * NP + MM + PP + k) * DD + c;
      x[off] = v; xbf[off] = f2bf(v);
    }
  }
}

// ---------------- edge construction ----------------
template <int KK>
__device__ inline void topk_insert(float v, int idx, float* bv, int* bi) {
  if (v <= bv[KK - 1]) return;
  int p = KK - 1;
  while (p > 0 && bv[p - 1] < v) { bv[p] = bv[p - 1]; bi[p] = bi[p - 1]; --p; }
  bv[p] = v; bi[p] = idx;
}

__global__ __launch_bounds__(256)
void build_edges_kernel(const float* __restrict__ x,
                        const float* __restrict__ gbias, const float* __restrict__ rbias,
                        int* __restrict__ esrc, int* __restrict__ edst, int* __restrict__ ecnt)
{
  const int b = blockIdx.x;
  const float* xs = x + (size_t)b * NP * DD;
  const float* g = xs;
  const float* r = xs + MM * DD;
  const float* a = xs + (MM + PP) * DD;   // already unit rows
  __shared__ int cnt;
  __shared__ float rinv[PP];
  if (threadIdx.x == 0) cnt = 0;
  __syncthreads();
  int* es = esrc + (size_t)b * EMAXE;
  int* ed = edst + (size_t)b * EMAXE;

  // fixed 24x24 grid 4-neighborhood (bidirectional)
  for (int c = threadIdx.x; c < MM; c += blockDim.x) {
    int i = c / GSZ, j = c % GSZ;
    if (j < GSZ - 1) { int p = atomicAdd(&cnt, 2); es[p] = c; ed[p] = c + 1; es[p+1] = c + 1; ed[p+1] = c; }
    if (i < GSZ - 1) { int p = atomicAdd(&cnt, 2); es[p] = c; ed[p] = c + GSZ; es[p+1] = c + GSZ; ed[p+1] = c; }
  }
  // inverse norms of region rows
  for (int i = threadIdx.x; i < PP; i += blockDim.x) {
    float s = 0.f;
    for (int c = 0; c < DD; ++c) { float v = r[i * DD + c]; s += v * v; }
    rinv[i] = 1.f / fmaxf(sqrtf(s), 1e-12f);
  }
  __syncthreads();

  // region-region cosine, top-5 > 0.5
  for (int i = threadIdx.x; i < PP; i += blockDim.x) {
    float bv[5]; int bi[5];
#pragma unroll
    for (int t = 0; t < 5; ++t) { bv[t] = -1e30f; bi[t] = 0; }
    for (int j = 0; j < PP; ++j) {
      if (j == i) continue;
      float s = 0.f;
      for (int c = 0; c < DD; ++c) s += r[i * DD + c] * r[j * DD + c];
      topk_insert<5>(s * rinv[i] * rinv[j], j, bv, bi);
    }
#pragma unroll
    for (int t = 0; t < 5; ++t)
      if (bv[t] > 0.5f) {
        int p = atomicAdd(&cnt, 2);
        es[p] = MM + i; ed[p] = MM + bi[t]; es[p+1] = MM + bi[t]; ed[p+1] = MM + i;
      }
  }
  // grid-cluster softmax corr, top-3 > 0.1
  const float bias = gbias[0] + rbias[0];
  for (int m = threadIdx.x; m < MM; m += blockDim.x) {
    float sc[KC]; float mx = -1e30f;
#pragma unroll
    for (int k = 0; k < KC; ++k) {
      float s = 0.f;
      for (int c = 0; c < DD; ++c) s += g[m * DD + c] * a[k * DD + c];
      sc[k] = s + bias; mx = fmaxf(mx, sc[k]);
    }
    float den = 0.f;
#pragma unroll
    for (int k = 0; k < KC; ++k) { sc[k] = __expf(sc[k] - mx); den += sc[k]; }
    float bv[3]; int bi[3];
#pragma unroll
    for (int t = 0; t < 3; ++t) { bv[t] = -1e30f; bi[t] = 0; }
#pragma unroll
    for (int k = 0; k < KC; ++k) topk_insert<3>(sc[k] / den, k, bv, bi);
#pragma unroll
    for (int t = 0; t < 3; ++t)
      if (bv[t] > 0.1f) {
        int p = atomicAdd(&cnt, 2);
        es[p] = m; ed[p] = MM + PP + bi[t]; es[p+1] = MM + PP + bi[t]; ed[p+1] = m;
      }
  }
  // region-cluster cosine, top-2 > 0.3
  for (int i = threadIdx.x; i < PP; i += blockDim.x) {
    float bv[2]; int bi[2];
#pragma unroll
    for (int t = 0; t < 2; ++t) { bv[t] = -1e30f; bi[t] = 0; }
#pragma unroll
    for (int k = 0; k < KC; ++k) {
      float s = 0.f;
      for (int c = 0; c < DD; ++c) s += r[i * DD + c] * a[k * DD + c];
      topk_insert<2>(s * rinv[i], k, bv, bi);
    }
#pragma unroll
    for (int t = 0; t < 2; ++t)
      if (bv[t] > 0.3f) {
        int p = atomicAdd(&cnt, 2);
        es[p] = MM + i; ed[p] = MM + PP + bi[t]; es[p+1] = MM + PP + bi[t]; ed[p+1] = MM + i;
      }
  }
  __syncthreads();
  if (threadIdx.x == 0) ecnt[b] = cnt;
}

// ---------------- edge attention ----------------
// pass A: per (edge, head): score = q[dst].k[src]/sqrt(DH); atomic max into smax
__global__ __launch_bounds__(256)
void att_max_kernel(const unsigned short* __restrict__ qkv,
                    const int* __restrict__ esrc, const int* __restrict__ edst,
                    const int* __restrict__ ecnt, unsigned* __restrict__ smax)
{
  const int e = blockIdx.x, b = blockIdx.y;
  if (e >= ecnt[b]) return;
  const int h = threadIdx.x >> 5, lane = threadIdx.x & 31;
  const int s = esrc[(size_t)b * EMAXE + e];
  const int d = edst[(size_t)b * EMAXE + e];
  const unsigned short* qk = qkv + (size_t)b * NP * 1024;
  float p = bf2f(qk[(size_t)d * 1024 + h * DHH + lane]) *
            bf2f(qk[(size_t)s * 1024 + DD + h * DHH + lane]);
  for (int o = 16; o; o >>= 1) p += __shfl_xor(p, o);
  if (lane == 0)
    atomicMax(&smax[((size_t)b * NP + d) * HH + h], fenc(p * 0.17677669529663687f));
}

// pass B: ex = exp(score - smax); denom += ex; hagg[dst] += ex * v[src]
__global__ __launch_bounds__(256)
void att_accum_kernel(const unsigned short* __restrict__ qkv,
                      const int* __restrict__ esrc, const int* __restrict__ edst,
                      const int* __restrict__ ecnt, const unsigned* __restrict__ smax,
                      float* __restrict__ denom, float* __restrict__ hagg)
{
  const int e = blockIdx.x, b = blockIdx.y;
  if (e >= ecnt[b]) return;
  const int h = threadIdx.x >> 5, lane = threadIdx.x & 31;
  const int s = esrc[(size_t)b * EMAXE + e];
  const int d = edst[(size_t)b * EMAXE + e];
  const unsigned short* qk = qkv + (size_t)b * NP * 1024;
  float p = bf2f(qk[(size_t)d * 1024 + h * DHH + lane]) *
            bf2f(qk[(size_t)s * 1024 + DD + h * DHH + lane]);
  for (int o = 16; o; o >>= 1) p += __shfl_xor(p, o);
  const float score = p * 0.17677669529663687f;
  float sm = fdec(smax[((size_t)b * NP + d) * HH + h]);
  if (sm < -5e29f) sm = 0.f;           // nodes with no valid in-edge
  const float ex = __expf(score - sm);
  const float vv = bf2f(qk[(size_t)s * 1024 + 2 * DD + h * DHH + lane]);
  atomicAdd(&hagg[((size_t)b * NP + d) * DD + h * DHH + lane], ex * vv);
  if (lane == 0) atomicAdd(&denom[((size_t)b * NP + d) * HH + h], ex);
}

// ---------------- residual + LayerNorm ----------------
__global__ __launch_bounds__(256)
void layer_finalize_kernel(float* __restrict__ x, unsigned short* __restrict__ xbf,
                           const float* __restrict__ hagg, const float* __restrict__ denom,
                           const unsigned short* __restrict__ qkv,
                           const float* __restrict__ ln_g, const float* __restrict__ ln_b)
{
  const int b = blockIdx.y, c = threadIdx.x;
  const size_t node = (size_t)b * NP + blockIdx.x;
  const float skip = bf2f(qkv[node * 1024 + 3 * DD + c]);
  const float den  = denom[node * HH + (c >> 5)];
  const float agg  = hagg[node * DD + c] / fmaxf(den, 1e-16f);
  const float val  = x[node * DD + c] + agg + skip;
  __shared__ float red[DD];
  red[c] = val; __syncthreads();
  for (int s = DD / 2; s > 0; s >>= 1) { if (c < s) red[c] += red[c + s]; __syncthreads(); }
  const float mu = red[0] / (float)DD;
  __syncthreads();
  const float dv = val - mu;
  red[c] = dv * dv; __syncthreads();
  for (int s = DD / 2; s > 0; s >>= 1) { if (c < s) red[c] += red[c + s]; __syncthreads(); }
  const float var = red[0] / (float)DD;
  const float y = dv * rsqrtf(var + 1e-5f) * ln_g[c] + ln_b[c];
  x[node * DD + c] = y;
  xbf[node * DD + c] = f2bf(y);
}

// ---------------- output scatter ----------------
__global__ void scatter_out_kernel(const float* __restrict__ xo, float* __restrict__ out) {
  const long total = (long)BB * NN * DD;
  long i = (long)blockIdx.x * blockDim.x + threadIdx.x;
  const long st = (long)gridDim.x * blockDim.x;
  const size_t O1 = (size_t)BB * MM * DD;
  const size_t O2 = O1 + (size_t)BB * PP * DD;
  for (; i < total; i += st) {
    const int c = (int)(i % DD);
    const long nn = i / DD;
    const int n = (int)(nn % NN);
    const int b = (int)(nn / NN);
    const float v = xo[((size_t)b * NP + n) * DD + c];
    size_t dst;
    if (n < MM)            dst = ((size_t)b * MM + n) * DD + c;
    else if (n < MM + PP)  dst = O1 + ((size_t)b * PP + (n - MM)) * DD + c;
    else                   dst = O2 + ((size_t)b * KC + (n - MM - PP)) * DD + c;
    out[dst] = v;
  }
}

// ---------------- host launch ----------------
extern "C" void kernel_launch(void* const* d_in, const int* in_sizes, int n_in,
                              void* d_out, int out_size, void* d_ws, size_t ws_size,
                              hipStream_t stream)
{
  (void)in_sizes; (void)n_in; (void)out_size; (void)ws_size;
  // params flattened: sorted dict keys (jax pytree order)
  const float* grid_f   = (const float*)d_in[0];
  const float* region_f = (const float*)d_in[1];
  const float* centers  = (const float*)d_in[2];
  const float* gbias    = (const float*)d_in[3];
  const float* gprojb   = (const float*)d_in[4];
  const float* gprojw   = (const float*)d_in[5];
  const float *bk[3], *bq[3], *bs[3], *bv[3], *lnb[3], *lng[3], *wk[3], *wq[3], *wsp[3], *wv[3];
  for (int i = 0; i < 3; ++i) {
    int base = 6 + 10 * i;
    bk[i]  = (const float*)d_in[base + 0];
    bq[i]  = (const float*)d_in[base + 1];
    bs[i]  = (const float*)d_in[base + 2];
    bv[i]  = (const float*)d_in[base + 3];
    lnb[i] = (const float*)d_in[base + 4];
    lng[i] = (const float*)d_in[base + 5];
    wk[i]  = (const float*)d_in[base + 6];
    wq[i]  = (const float*)d_in[base + 7];
    wsp[i] = (const float*)d_in[base + 8];
    wv[i]  = (const float*)d_in[base + 9];
  }
  const float* out_b  = (const float*)d_in[36];
  const float* out_w  = (const float*)d_in[37];
  const float* r_bar  = (const float*)d_in[38];
  const float* rbias  = (const float*)d_in[39];
  const float* rprojb = (const float*)d_in[40];
  const float* rprojw = (const float*)d_in[41];

  char* ws = (char*)d_ws;
  size_t off = 0;
  auto take = [&](size_t bytes) { size_t o = off; off = (off + bytes + 255) & ~(size_t)255; return (void*)(ws + o); };
  const size_t XE = (size_t)BB * NP * DD;
  float*          X    = (float*)         take(XE * 4);
  unsigned short* XBF  = (unsigned short*)take(XE * 2);
  unsigned short* QKV  = (unsigned short*)take((size_t)BB * NP * 1024 * 2);
  float*          HAGG = (float*)         take(XE * 4);
  unsigned*       SMAX = (unsigned*)      take((size_t)BB * NP * HH * 4);
  float*          DEN  = (float*)         take((size_t)BB * NP * HH * 4);
  int*            ESRC = (int*)           take((size_t)BB * EMAXE * 4);
  int*            EDST = (int*)           take((size_t)BB * EMAXE * 4);
  int*            ECNT = (int*)           take((size_t)BB * 4);
  unsigned short* GFB  = (unsigned short*)take((size_t)BB * MM * 128 * 2);
  unsigned short* RFB  = (unsigned short*)take((size_t)BB * PP * 512 * 2);
  unsigned short* WG   = (unsigned short*)take((size_t)256 * 128 * 2);
  unsigned short* WR   = (unsigned short*)take((size_t)256 * 512 * 2);
  unsigned short* WL   = (unsigned short*)take((size_t)3 * 1024 * 256 * 2);
  float*          BL   = (float*)         take((size_t)3 * 1024 * 4);
  unsigned short* WO   = (unsigned short*)take((size_t)256 * 256 * 2);

  // 1) convert inputs to bf16
  f2bf_kernel<<<4096, 256, 0, stream>>>(grid_f,   GFB, (long)BB * MM * 128);
  f2bf_kernel<<<4096, 256, 0, stream>>>(region_f, RFB, (long)BB * PP * 512);
  // 2) pack weights (transposed, bf16)
  pack_wt_kernel<<<(128 * 256 + 255) / 256, 256, 0, stream>>>(gprojw, WG, 128, 256);
  pack_wt_kernel<<<(512 * 256 + 255) / 256, 256, 0, stream>>>(rprojw, WR, 512, 256);
  pack_wt_kernel<<<(256 * 256 + 255) / 256, 256, 0, stream>>>(out_w,  WO, 256, 256);
  for (int i = 0; i < 3; ++i) {
    unsigned short* WLi = WL + (size_t)i * 1024 * 256;
    pack_wt_kernel<<<256, 256, 0, stream>>>(wq[i],  WLi + 0 * 65536, 256, 256);
    pack_wt_kernel<<<256, 256, 0, stream>>>(wk[i],  WLi + 1 * 65536, 256, 256);
    pack_wt_kernel<<<256, 256, 0, stream>>>(wv[i],  WLi + 2 * 65536, 256, 256);
    pack_wt_kernel<<<256, 256, 0, stream>>>(wsp[i], WLi + 3 * 65536, 256, 256);
    float* BLi = BL + (size_t)i * 1024;
    copy_f32_kernel<<<1, 256, 0, stream>>>(bq[i], BLi + 0,   256);
    copy_f32_kernel<<<1, 256, 0, stream>>>(bk[i], BLi + 256, 256);
    copy_f32_kernel<<<1, 256, 0, stream>>>(bv[i], BLi + 512, 256);
    copy_f32_kernel<<<1, 256, 0, stream>>>(bs[i], BLi + 768, 256);
  }
  zero_pad_kernel<<<(BB * (NP - NN) * 256 + 255) / 256, 256, 0, stream>>>(X, XBF);

  // 3) projections: gh = grid_f @ Wg + b ; rh = region_f @ Wr + b  (rows 0..639 of x)
  {
    int blks = (MM / 32) * (DD / 64);
    gemm_bf16_wmma<<<dim3((blks + 7) / 8, BB), 256, 0, stream>>>(
        GFB, (long)MM * 128, 128, WG, 128, gprojb,
        X, (long)NP * DD, XBF, (long)NP * DD, DD, MM / 32, DD / 64, 128);
  }
  {
    int blks = (PP / 32) * (DD / 64);
    gemm_bf16_wmma<<<dim3((blks + 7) / 8, BB), 256, 0, stream>>>(
        RFB, (long)PP * 512, 512, WR, 512, rprojb,
        X + (size_t)MM * DD, (long)NP * DD, XBF + (size_t)MM * DD, (long)NP * DD,
        DD, PP / 32, DD / 64, 512);
  }
  // 4) cluster aggregation -> rows 640..649
  cluster_agg_kernel<<<BB, 256, 0, stream>>>(X, XBF, centers, r_bar, gbias, rbias);
  // 5) dynamic edges
  build_edges_kernel<<<BB, 256, 0, stream>>>(X, gbias, rbias, ESRC, EDST, ECNT);

  // 6) transformer conv layers
  for (int i = 0; i < 3; ++i) {
    init_att_kernel<<<16384, 256, 0, stream>>>(SMAX, DEN, HAGG);
    int blks = (NP / 32) * (1024 / 64);
    gemm_bf16_wmma<<<dim3((blks + 7) / 8, BB), 256, 0, stream>>>(
        XBF, (long)NP * DD, DD, WL + (size_t)i * 1024 * 256, DD, BL + (size_t)i * 1024,
        nullptr, 0, QKV, (long)NP * 1024, 1024, NP / 32, 1024 / 64, DD);
    att_max_kernel<<<dim3(EMAXE, BB), 256, 0, stream>>>(QKV, ESRC, EDST, ECNT, SMAX);
    att_accum_kernel<<<dim3(EMAXE, BB), 256, 0, stream>>>(QKV, ESRC, EDST, ECNT, SMAX, DEN, HAGG);
    layer_finalize_kernel<<<dim3(NP, BB), 256, 0, stream>>>(X, XBF, HAGG, DEN, QKV, lng[i], lnb[i]);
  }

  // 7) output projection + scatter into tuple-ordered d_out
  {
    int blks = (NP / 32) * (DD / 64);
    gemm_bf16_wmma<<<dim3((blks + 7) / 8, BB), 256, 0, stream>>>(
        XBF, (long)NP * DD, DD, WO, DD, out_b,
        HAGG, (long)NP * DD, nullptr, 0, DD, NP / 32, DD / 64, DD);
  }
  scatter_out_kernel<<<4096, 256, 0, stream>>>(HAGG, (float*)d_out);
}
